// CGF_38113539785332
// MI455X (gfx1250) — matrix operand order repaired and checked by reading, not compiled, not run on hardware
//
#include <hip/hip_runtime.h>
#include <stdint.h>

// ---------------------------------------------------------------------------
// Tree-GRU on MI455X (gfx1250): bf16 WMMA GEMMs (fp32 accumulate) + fp32 gates
// B weight panels staged to LDS via gfx1250 async-to-LDS copies.
// ---------------------------------------------------------------------------

typedef __attribute__((ext_vector_type(16))) __bf16   v16bf;
typedef __attribute__((ext_vector_type(8)))  float    v8f;
typedef __attribute__((ext_vector_type(8)))  uint32_t v8u;
typedef __attribute__((ext_vector_type(4)))  uint32_t v4u;

#define TDIM   384            // GRU dim
#define GDIM   1152           // 3*DIM gate dim
#define NLEAF  32768          // 8^5
#define MAXM   4096           // parents at deepest level
#define CTILES 72             // GDIM/16 column tiles
#define KBLKS  12             // TDIM/32 k-blocks

__device__ __forceinline__ uint16_t f2bf(float f) {
  uint32_t u = __float_as_uint(f);
  u += 0x7FFFu + ((u >> 16) & 1u);      // round-to-nearest-even
  return (uint16_t)(u >> 16);
}

// Pack W[1152][384] fp32 -> bf16 WMMA-B panels: layout ((ct*12+kb)*32+lane)*16+e
// lane<16: N=lane, K=kb*32+e ; lane>=16: N=lane-16, K=kb*32+16+e
__global__ void pack_w(const float* __restrict__ W, uint16_t* __restrict__ Bp) {
  int idx = blockIdx.x * blockDim.x + threadIdx.x;
  if (idx >= CTILES * KBLKS * 32 * 16) return;
  int e    = idx & 15;
  int r    = idx >> 4;
  int lane = r & 31;
  int r2   = r >> 5;
  int kb   = r2 % KBLKS;
  int ct   = r2 / KBLKS;
  int n = ct * 16 + (lane & 15);
  int K = kb * 32 + (lane >> 4) * 16 + e;
  Bp[idx] = f2bf(W[(size_t)n * TDIM + K]);
}

__global__ void gather_embed(const int* __restrict__ tok,
                             const float* __restrict__ table,
                             uint16_t* __restrict__ Xbf) {
  int idx = blockIdx.x * blockDim.x + threadIdx.x;
  if (idx >= NLEAF * TDIM) return;
  int row = idx / TDIM;
  int d   = idx - row * TDIM;
  int t   = tok[row];
  Xbf[idx] = f2bf(table[(size_t)t * TDIM + d]);
}

__global__ void zero_h(float* __restrict__ Hf, uint16_t* __restrict__ Hbf, int n) {
  int idx = blockIdx.x * blockDim.x + threadIdx.x;
  if (idx < n) { Hf[idx] = 0.0f; Hbf[idx] = 0; }
}

// C[M,1152] = A[M,384](bf16, row stride a_stride elems) @ Wpanel^T  (fp32 out)
// Block = 4 waves; block stages its 48KB B panel (4 column tiles x full K)
// into LDS once via global_load_async_to_lds_b128, then each wave computes one
// 16-row M tile x four 16-col tiles (48 v_wmma per wave, B via ds_load_b128).
__global__ __launch_bounds__(128) void gemm_bf16(
    const uint16_t* __restrict__ A, int a_stride,
    const uint16_t* __restrict__ Bp, float* __restrict__ C, int M) {
  __shared__ __align__(16) uint16_t Bs[4 * KBLKS * 32 * 16];   // 49152 B

  const int wave = threadIdx.x >> 5;
  const int lane = threadIdx.x & 31;
  const int ct0  = blockIdx.x * 4;

  // ---- async copy of the contiguous 48KB B panel into LDS (all threads) ----
  {
    const uint16_t* src = Bp + (size_t)ct0 * KBLKS * 32 * 16;
    uint32_t lbase = (uint32_t)(uintptr_t)&Bs[0];   // low 32 bits = LDS offset
    #pragma unroll
    for (int i = 0; i < 24; ++i) {                  // 3072 x 16B / 128 threads
      int idx16 = threadIdx.x + i * 128;            // 16-byte chunk index
      const uint16_t* g = src + (size_t)idx16 * 8;  // 8 elems = 16 bytes
      uint32_t l = lbase + (uint32_t)idx16 * 16;
      asm volatile("global_load_async_to_lds_b128 %0, %1, off"
                   :: "v"(l), "v"(g) : "memory");
    }
    asm volatile("s_wait_asynccnt 0x0" ::: "memory");
  }
  __syncthreads();

  const int mtiles = (M + 15) >> 4;
  const int mtile  = blockIdx.y * 4 + wave;
  if (mtile >= mtiles) return;

  const int m  = lane & 15;
  const int hs = lane >> 4;                 // lane-group: which K half
  int row = mtile * 16 + m;
  if (row >= M) row = 0;                    // clamp; those C rows never stored
  const uint16_t* arow = A + (size_t)row * a_stride;
  const uint16_t* bbase = &Bs[lane * 16];

  v8f acc[4] = {};
  #pragma unroll
  for (int kb = 0; kb < KBLKS; ++kb) {
    // A 16x32 bf16 layout: this lane holds K = hs*16 + [0..15] (matches B panel)
    const int k0 = kb * 32 + hs * 16;
    v4u alo = *(const v4u*)(arow + k0);
    v4u ahi = *(const v4u*)(arow + k0 + 8);
    v8u au;
    au[0]=alo[0]; au[1]=alo[1]; au[2]=alo[2]; au[3]=alo[3];
    au[4]=ahi[0]; au[5]=ahi[1]; au[6]=ahi[2]; au[7]=ahi[3];
    v16bf av = __builtin_bit_cast(v16bf, au);
    #pragma unroll
    for (int c = 0; c < 4; ++c) {
      const uint16_t* bp = bbase + ((size_t)c * KBLKS + kb) * 512;
      v4u blo = *(const v4u*)(bp);
      v4u bhi = *(const v4u*)(bp + 8);
      v8u bu;
      bu[0]=blo[0]; bu[1]=blo[1]; bu[2]=blo[2]; bu[3]=blo[3];
      bu[4]=bhi[0]; bu[5]=bhi[1]; bu[6]=bhi[2]; bu[7]=bhi[3];
      v16bf bv = __builtin_bit_cast(v16bf, bu);
      acc[c] = __builtin_amdgcn_wmma_f32_16x16x32_bf16(
          false, av, false, bv, (short)0, acc[c], false, false);
    }
  }
  // C/D layout: VGPR j -> M = hs*8 + j ; N = lane&15
  #pragma unroll
  for (int c = 0; c < 4; ++c) {
    const int col = (ct0 + c) * 16 + m;
    #pragma unroll
    for (int j = 0; j < 8; ++j) {
      const int r = mtile * 16 + hs * 8 + j;
      if (r < M) C[(size_t)r * GDIM + col] = acc[c][j];
    }
  }
}

// GRU gates in fp32: r,z,n order, biases added here (b_hh_n is inside r*(...))
__global__ void gru_pointwise(const float* __restrict__ GI, const float* __restrict__ GH,
                              const float* __restrict__ b_ih, const float* __restrict__ b_hh,
                              float* __restrict__ Hf, uint16_t* __restrict__ Hbf,
                              float* __restrict__ Xout, int M, int t) {
  int idx = blockIdx.x * blockDim.x + threadIdx.x;
  if (idx >= M * TDIM) return;
  int p = idx / TDIM;
  int d = idx - p * TDIM;
  size_t g = (size_t)p * GDIM;
  float ir = GI[g + d]            + b_ih[d];
  float iz = GI[g + TDIM + d]     + b_ih[TDIM + d];
  float in_= GI[g + 2 * TDIM + d] + b_ih[2 * TDIM + d];
  float hr = GH[g + d]            + b_hh[d];
  float hz = GH[g + TDIM + d]     + b_hh[TDIM + d];
  float hn = GH[g + 2 * TDIM + d] + b_hh[2 * TDIM + d];
  float r = 1.0f / (1.0f + __expf(-(ir + hr)));
  float z = 1.0f / (1.0f + __expf(-(iz + hz)));
  float n = tanhf(in_ + r * hn);
  float h = Hf[idx];
  float hnew = (1.0f - z) * n + z * h;
  Hf[idx]  = hnew;
  Hbf[idx] = f2bf(hnew);
  Xout[idx] = (t == 0 ? 0.0f : Xout[idx]) + hnew;
}

// Level transition: x_next = Xout/8 (bf16) ; h0_next = mean of 8 child h_last
__global__ void finish_level(const float* __restrict__ Xout, uint16_t* __restrict__ Xbf,
                             const float* __restrict__ Hcur, float* __restrict__ Hnext,
                             uint16_t* __restrict__ Hbf, int M) {
  int idx = blockIdx.x * blockDim.x + threadIdx.x;
  if (idx >= M * TDIM) return;
  float x = Xout[idx] * 0.125f;
  Xbf[idx] = f2bf(x);
  int p = idx / TDIM;
  int d = idx - p * TDIM;
  int Mn = M >> 3;
  if (p < Mn) {
    float s = 0.0f;
    #pragma unroll
    for (int c = 0; c < 8; ++c) s += Hcur[(size_t)(p * 8 + c) * TDIM + d];
    float h0 = s * 0.125f;
    Hnext[idx] = h0;
    Hbf[idx]   = f2bf(h0);
  }
}

__global__ void store_out(const float* __restrict__ Xout, float* __restrict__ out) {
  int d = blockIdx.x * blockDim.x + threadIdx.x;
  if (d < TDIM) out[d] = Xout[d] * 0.125f;
}

extern "C" void kernel_launch(void* const* d_in, const int* in_sizes, int n_in,
                              void* d_out, int out_size, void* d_ws, size_t ws_size,
                              hipStream_t stream) {
  const int*   tok   = (const int*)d_in[0];
  const float* table = (const float*)d_in[1];
  const float* w_ih  = (const float*)d_in[2];
  const float* w_hh  = (const float*)d_in[3];
  const float* b_ih  = (const float*)d_in[4];
  const float* b_hh  = (const float*)d_in[5];
  float* out = (float*)d_out;

  char* base = (char*)d_ws;
  size_t off = 0;
  auto alloc = [&](size_t b) { void* p = base + off; off += (b + 255) & ~(size_t)255; return p; };
  uint16_t* BpIH = (uint16_t*)alloc((size_t)GDIM * TDIM * 2);   // 0.88 MB
  uint16_t* BpHH = (uint16_t*)alloc((size_t)GDIM * TDIM * 2);   // 0.88 MB
  uint16_t* Xbf  = (uint16_t*)alloc((size_t)NLEAF * TDIM * 2);  // 25 MB
  uint16_t* Hbf  = (uint16_t*)alloc((size_t)MAXM * TDIM * 2);   // 3.1 MB
  float*    HfA  = (float*)alloc((size_t)MAXM * TDIM * 4);      // 6.3 MB
  float*    HfB  = (float*)alloc((size_t)MAXM * TDIM * 4);      // 6.3 MB
  float*    GI   = (float*)alloc((size_t)MAXM * GDIM * 4);      // 18.9 MB
  float*    GH   = (float*)alloc((size_t)MAXM * GDIM * 4);      // 18.9 MB
  float*    Xout = (float*)alloc((size_t)MAXM * TDIM * 4);      // 6.3 MB

  const int packN = CTILES * KBLKS * 32 * 16;
  pack_w<<<dim3((packN + 255) / 256), dim3(256), 0, stream>>>(w_ih, BpIH);
  pack_w<<<dim3((packN + 255) / 256), dim3(256), 0, stream>>>(w_hh, BpHH);
  gather_embed<<<dim3((NLEAF * TDIM + 255) / 256), dim3(256), 0, stream>>>(tok, table, Xbf);
  zero_h<<<dim3((MAXM * TDIM + 255) / 256), dim3(256), 0, stream>>>(HfA, Hbf, MAXM * TDIM);

  float* Hcur = HfA;
  float* Hnext = HfB;
  int M = MAXM;                                // parents at level 5
  for (int level = 5; level >= 1; --level) {
    int mtiles = (M + 15) >> 4;
    dim3 ggrid(CTILES / 4, (mtiles + 3) / 4);
    for (int t = 0; t < 8; ++t) {
      // reversed children: step t reads child (7-t) of each parent
      const uint16_t* Ax = Xbf + (size_t)(7 - t) * TDIM;
      gemm_bf16<<<ggrid, dim3(128), 0, stream>>>(Ax, 8 * TDIM, BpIH, GI, M);
      gemm_bf16<<<ggrid, dim3(128), 0, stream>>>(Hbf, TDIM, BpHH, GH, M);
      gru_pointwise<<<dim3((M * TDIM + 255) / 256), dim3(256), 0, stream>>>(
          GI, GH, b_ih, b_hh, Hcur, Hbf, Xout, M, t);
    }
    if (level > 1) {
      finish_level<<<dim3((M * TDIM + 255) / 256), dim3(256), 0, stream>>>(
          Xout, Xbf, Hcur, Hnext, Hbf, M);
      float* tmp = Hcur; Hcur = Hnext; Hnext = tmp;
      M >>= 3;
    } else {
      store_out<<<dim3(2), dim3(256), 0, stream>>>(Xout, out);
    }
  }
  (void)in_sizes; (void)n_in; (void)out_size; (void)ws_size;
}
